// ModelLikeAnirbans_55353538511387
// MI455X (gfx1250) — matrix operand order repaired and checked by reading, not compiled, not run on hardware
//
#include <hip/hip_runtime.h>
#include <hip/hip_bf16.h>
#include <cstdint>
#include <cstddef>

typedef __attribute__((ext_vector_type(16))) __bf16 v16bf;
typedef __attribute__((ext_vector_type(8)))  __bf16 v8bf;
typedef __attribute__((ext_vector_type(8)))  float  v8f;

#define GRID_N   512
#define NN       (GRID_N * GRID_N)      /* 262144 nodes  */
#define NE       (4 * NN)               /* 1048576 edges */
#define CDIM     32
#define COUT     64
#define INNER    510
#define NKEEP    (INNER * INNER)        /* 260100 interior nodes */
#define H1D      256
#define H2D      256
#define OUTD     16

// ---------------------------------------------------------------- WMMA helpers

static __device__ __forceinline__ v8f wmma_bf16(v16bf a, v16bf b, v8f c) {
  // D = A(16x32 bf16) * B(32x16 bf16) + C(16x16 f32)
  return __builtin_amdgcn_wmma_f32_16x16x32_bf16(false, a, false, b,
                                                 (short)0, c, false, false);
}

// A-fragment per ISA 16-bit A 16x32 layout: lane half h, element e:
//   e 0..7  -> K = kbase + h*8 + e        (first 16 bytes)
//   e 8..15 -> K = kbase + 16 + h*8 + e'  (second 16 bytes)
static __device__ __forceinline__ v16bf a_frag(const __bf16* row, int kbase, int h) {
  union { v16bf v; v8bf p[2]; } u;
  u.p[0] = *reinterpret_cast<const v8bf*>(row + kbase + h * 8);
  u.p[1] = *reinterpret_cast<const v8bf*>(row + kbase + 16 + h * 8);
  return u.v;
}

// B-fragment from packed weights: 32 bytes contiguous per lane.
static __device__ __forceinline__ v16bf b_frag(const __bf16* pk, int kt, int nt,
                                               int ntiles, int lane) {
  return *reinterpret_cast<const v16bf*>(
      pk + (((size_t)(kt * ntiles + nt) * 32 + lane) << 4));
}

// ---------------------------------------------------------------- prep kernels

__global__ void zero_f4_kernel(float4* p, int n4) {
  int i = blockIdx.x * blockDim.x + threadIdx.x;
  if (i < n4) p[i] = make_float4(0.f, 0.f, 0.f, 0.f);
}

__global__ void cvt_bf16_kernel(const float* s, __bf16* d, int n) {
  int i = blockIdx.x * blockDim.x + threadIdx.x;
  if (i < n) d[i] = (__bf16)s[i];
}

// Pack KxN row-major f32 weights into the B-fragment layout:
// dst[((kt*ntiles+nt)*32 + lane)*16 + e] = W[kt*32 + (e<8 ? (lane/16)*8 + e : 16 + (lane/16)*8 + e-8)][nt*16 + lane%16]
__global__ void pack_weight_kernel(const float* src, __bf16* dst, int K, int N) {
  int lane   = threadIdx.x & 31;
  int waveId = (blockIdx.x * blockDim.x + threadIdx.x) >> 5;
  int ktiles = K >> 5, ntiles = N >> 4;
  if (waveId >= ktiles * ntiles) return;
  int kt = waveId / ntiles, nt = waveId % ntiles;
  int h = lane >> 4, nn = lane & 15;
  __bf16* d = dst + ((size_t)waveId * 32 + lane) * 16;
#pragma unroll
  for (int e = 0; e < 16; ++e) {
    int kk = kt * 32 + (e < 8 ? h * 8 + e : 16 + h * 8 + (e - 8));
    d[e] = (__bf16)src[(size_t)kk * N + nt * 16 + nn];
  }
}

// ---------------------------------------------------------------- edge kernel
// 16 edges per wave: t=[x_i|x_j] -> relu(t@Wm1+b) @ Wm2 + b ; msg = (h*(xi-xj))@Wlin2+b
// scatter-add into agg[dst].
__global__ void edge_kernel(const float* __restrict__ convs,
                            const int*   __restrict__ edges,
                            const __bf16* __restrict__ pkWm1,
                            const __bf16* __restrict__ pkWm2,
                            const __bf16* __restrict__ pkWlin2,
                            const float* __restrict__ b_m1,
                            const float* __restrict__ b_m2,
                            const float* __restrict__ b_lin2,
                            float* __restrict__ agg) {
  constexpr int WAVES = 4;
  __shared__ __align__(16) __bf16 tA[WAVES][16][72];    // [x_i | x_j] bf16
  __shared__ __align__(16) float  xF[WAVES][16][64];    // [x_i | x_j] f32
  __shared__ __align__(16) __bf16 hT[WAVES][16][136];   // hidden 128
  __shared__ __align__(16) __bf16 mT[WAVES][16][40];    // h*(xi-xj)  32
  __shared__ int dIdx[WAVES][16];

  const int lane = threadIdx.x & 31;
  const int wv   = threadIdx.x >> 5;
  const int m    = lane & 15;   // row within tile (also N for C-tiles)
  const int h    = lane >> 4;   // lane half
  const int e0   = (blockIdx.x * WAVES + wv) * 16;

  // ---- gather: lanes 0-15 fetch x_i (dst rows), lanes 16-31 fetch x_j (src rows)
  {
    int e = e0 + m;
    int node;
    if (h == 0) { node = edges[NE + e]; dIdx[wv][m] = node; }  // dst
    else        { node = edges[e]; }                            // src
    const float4* rowp = reinterpret_cast<const float4*>(convs + (size_t)node * CDIM);
    int cb = h * 32;
#pragma unroll
    for (int q = 0; q < 8; ++q) {
      float4 f = rowp[q];
      int c = cb + q * 4;
      xF[wv][m][c + 0] = f.x; xF[wv][m][c + 1] = f.y;
      xF[wv][m][c + 2] = f.z; xF[wv][m][c + 3] = f.w;
      tA[wv][m][c + 0] = (__bf16)f.x; tA[wv][m][c + 1] = (__bf16)f.y;
      tA[wv][m][c + 2] = (__bf16)f.z; tA[wv][m][c + 3] = (__bf16)f.w;
    }
  }
  __syncthreads();

  // ---- GEMM1: (16x64) @ Wm1(64x128) -> relu -> hT
  for (int nt = 0; nt < 8; ++nt) {
    v8f acc = {};
#pragma unroll
    for (int kt = 0; kt < 2; ++kt) {
      v16bf a = a_frag(&tA[wv][m][0], kt * 32, h);
      v16bf b = b_frag(pkWm1, kt, nt, 8, lane);
      acc = wmma_bf16(a, b, acc);
    }
    int col = nt * 16 + m;
    float bias = b_m1[col];
#pragma unroll
    for (int v = 0; v < 8; ++v) {
      float val = acc[v] + bias;
      hT[wv][v + 8 * h][col] = (__bf16)(val > 0.f ? val : 0.f);
    }
  }
  __syncthreads();

  // ---- GEMM2: (16x128) @ Wm2(128x32) + b, then * (x_i - x_j) -> mT
  for (int nt = 0; nt < 2; ++nt) {
    v8f acc = {};
#pragma unroll
    for (int kt = 0; kt < 4; ++kt) {
      v16bf a = a_frag(&hT[wv][m][0], kt * 32, h);
      v16bf b = b_frag(pkWm2, kt, nt, 2, lane);
      acc = wmma_bf16(a, b, acc);
    }
    int col = nt * 16 + m;
    float bias = b_m2[col];
#pragma unroll
    for (int v = 0; v < 8; ++v) {
      int M = v + 8 * h;
      float hv = acc[v] + bias;
      float df = xF[wv][M][col] - xF[wv][M][32 + col];
      mT[wv][M][col] = (__bf16)(hv * df);
    }
  }
  __syncthreads();

  // ---- GEMM3: (16x32) @ Wlin2(32x64) + b_lin2 -> atomic scatter into agg[dst]
  for (int nt = 0; nt < 4; ++nt) {
    v8f acc = {};
    v16bf a = a_frag(&mT[wv][m][0], 0, h);
    v16bf b = b_frag(pkWlin2, 0, nt, 4, lane);
    acc = wmma_bf16(a, b, acc);
    int col = nt * 16 + m;
    float bias = b_lin2[col];
#pragma unroll
    for (int v = 0; v < 8; ++v) {
      int M = v + 8 * h;
      atomicAdd(&agg[(size_t)dIdx[wv][M] * COUT + col], acc[v] + bias);
    }
  }
}

// ---------------------------------------------------------------- node + MLP
// 16 interior rows per wave: preconv = convs@Wlin1 + b + agg; x=[feat|preconv];
// relu(x@W1+b1); relu(@W2+b2); @W3+b3 -> out.
__global__ void node_mlp_kernel(const float* __restrict__ features,
                                const __bf16* __restrict__ convsbf,
                                const float* __restrict__ agg,
                                const __bf16* __restrict__ pkWlin1,
                                const float* __restrict__ b_lin1,
                                const __bf16* __restrict__ pkW1,
                                const float* __restrict__ b1,
                                const __bf16* __restrict__ pkW2,
                                const float* __restrict__ b2,
                                const __bf16* __restrict__ pkW3,
                                const float* __restrict__ b3,
                                float* __restrict__ out) {
  constexpr int WAVES = 2;
  __shared__ __align__(16) __bf16 xT[WAVES][16][136];   // [features | preconv]
  __shared__ __align__(16) __bf16 h1T[WAVES][16][264];
  __shared__ __align__(16) __bf16 h2T[WAVES][16][264];
  __shared__ int nIdx[WAVES][16];
  __shared__ int rIdx[WAVES][16];

  const int lane = threadIdx.x & 31;
  const int wv   = threadIdx.x >> 5;
  const int m    = lane & 15;
  const int h    = lane >> 4;
  const int tile = blockIdx.x * WAVES + wv;

  if (h == 0) {
    int r = tile * 16 + m;            // unclamped (store guard)
    rIdx[wv][m] = r;
    int rc = r < NKEEP ? r : NKEEP - 1;
    nIdx[wv][m] = (rc / INNER + 1) * GRID_N + (rc % INNER + 1);
  }
  __syncthreads();

  // ---- stage features (cols 0..63): lane handles row m, half h of the 64 cols
  {
    int r  = rIdx[wv][m];
    int rc = r < NKEEP ? r : NKEEP - 1;
    const float4* rowp = reinterpret_cast<const float4*>(features + (size_t)rc * 64);
    int cb = h * 32;
#pragma unroll
    for (int q = 0; q < 8; ++q) {
      float4 f = rowp[(h * 32) / 4 + q];
      int c = cb + q * 4;
      xT[wv][m][c + 0] = (__bf16)f.x; xT[wv][m][c + 1] = (__bf16)f.y;
      xT[wv][m][c + 2] = (__bf16)f.z; xT[wv][m][c + 3] = (__bf16)f.w;
    }
  }

  // ---- preconv: convs(16x32) @ Wlin1(32x64) + b_lin1 + agg -> cols 64..127
  {
    const __bf16* rowp = convsbf + (size_t)nIdx[wv][m] * CDIM;
    v16bf a = a_frag(rowp, 0, h);
    for (int nt = 0; nt < 4; ++nt) {
      v8f acc = {};
      v16bf b = b_frag(pkWlin1, 0, nt, 4, lane);
      acc = wmma_bf16(a, b, acc);
      int col = nt * 16 + m;
      float bias = b_lin1[col];
#pragma unroll
      for (int v = 0; v < 8; ++v) {
        int M = v + 8 * h;
        float val = acc[v] + bias + agg[(size_t)nIdx[wv][M] * COUT + col];
        xT[wv][M][64 + col] = (__bf16)val;
      }
    }
  }
  __syncthreads();

  // ---- layer 1: (16x128) @ W1(128x256) -> relu -> h1T
  for (int nt = 0; nt < 16; ++nt) {
    v8f acc = {};
#pragma unroll
    for (int kt = 0; kt < 4; ++kt) {
      v16bf a = a_frag(&xT[wv][m][0], kt * 32, h);
      v16bf b = b_frag(pkW1, kt, nt, 16, lane);
      acc = wmma_bf16(a, b, acc);
    }
    int col = nt * 16 + m;
    float bias = b1[col];
#pragma unroll
    for (int v = 0; v < 8; ++v) {
      float val = acc[v] + bias;
      h1T[wv][v + 8 * h][col] = (__bf16)(val > 0.f ? val : 0.f);
    }
  }
  __syncthreads();

  // ---- layer 2: (16x256) @ W2(256x256) -> relu -> h2T
  for (int nt = 0; nt < 16; ++nt) {
    v8f acc = {};
#pragma unroll
    for (int kt = 0; kt < 8; ++kt) {
      v16bf a = a_frag(&h1T[wv][m][0], kt * 32, h);
      v16bf b = b_frag(pkW2, kt, nt, 16, lane);
      acc = wmma_bf16(a, b, acc);
    }
    int col = nt * 16 + m;
    float bias = b2[col];
#pragma unroll
    for (int v = 0; v < 8; ++v) {
      float val = acc[v] + bias;
      h2T[wv][v + 8 * h][col] = (__bf16)(val > 0.f ? val : 0.f);
    }
  }
  __syncthreads();

  // ---- layer 3: (16x256) @ W3(256x16) + b3 -> out (f32)
  {
    v8f acc = {};
#pragma unroll
    for (int kt = 0; kt < 8; ++kt) {
      v16bf a = a_frag(&h2T[wv][m][0], kt * 32, h);
      v16bf b = b_frag(pkW3, kt, 0, 1, lane);
      acc = wmma_bf16(a, b, acc);
    }
    float bias = b3[m];
#pragma unroll
    for (int v = 0; v < 8; ++v) {
      int M = v + 8 * h;
      int r = rIdx[wv][M];
      if (r < NKEEP) out[(size_t)r * OUTD + m] = acc[v] + bias;
    }
  }
}

// ---------------------------------------------------------------- launch

extern "C" void kernel_launch(void* const* d_in, const int* in_sizes, int n_in,
                              void* d_out, int out_size, void* d_ws, size_t ws_size,
                              hipStream_t stream) {
  const float* convs    = (const float*)d_in[0];
  const float* features = (const float*)d_in[1];
  const int*   edges    = (const int*)d_in[2];
  // d_in[3] = weights (unused by reference), d_in[4] = coords (static halo mask)
  const float* W_lin1 = (const float*)d_in[5];
  const float* b_lin1 = (const float*)d_in[6];
  const float* W_lin2 = (const float*)d_in[7];
  const float* b_lin2 = (const float*)d_in[8];
  const float* W_m1   = (const float*)d_in[9];
  const float* b_m1   = (const float*)d_in[10];
  const float* W_m2   = (const float*)d_in[11];
  const float* b_m2   = (const float*)d_in[12];
  const float* W1     = (const float*)d_in[13];
  const float* b1     = (const float*)d_in[14];
  const float* W2     = (const float*)d_in[15];
  const float* b2     = (const float*)d_in[16];
  const float* W3     = (const float*)d_in[17];
  const float* b3     = (const float*)d_in[18];
  float* out = (float*)d_out;

  // workspace layout (needs ~84.2 MB)
  char* ws = (char*)d_ws;
  float*  agg     = (float*)ws;                                  // 262144*64 f32 = 64 MiB
  __bf16* convsbf = (__bf16*)(ws + (size_t)67108864);            // 262144*32 bf16 = 16 MiB
  __bf16* pk      = (__bf16*)(ws + (size_t)83886080);            // packed weights
  __bf16* pkWm1   = pk;               // 64x128  ->  8192 el
  __bf16* pkWm2   = pk + 8192;        // 128x32  ->  4096 el
  __bf16* pkWlin2 = pk + 12288;       // 32x64   ->  2048 el
  __bf16* pkWlin1 = pk + 14336;       // 32x64   ->  2048 el
  __bf16* pkW1    = pk + 16384;       // 128x256 -> 32768 el
  __bf16* pkW2    = pk + 49152;       // 256x256 -> 65536 el
  __bf16* pkW3    = pk + 114688;      // 256x16  ->  4096 el

  // 1) zero the aggregation buffer (fresh every call, deterministic)
  {
    int n4 = (NN * COUT) / 4;
    zero_f4_kernel<<<n4 / 256, 256, 0, stream>>>((float4*)agg, n4);
  }
  // 2) convs -> bf16
  {
    int n = NN * CDIM;
    cvt_bf16_kernel<<<n / 256, 256, 0, stream>>>(convs, convsbf, n);
  }
  // 3) pack weights into WMMA B-fragment layout
  {
    auto packs = [&](const float* s, __bf16* d, int K, int N) {
      int waves = (K >> 5) * (N >> 4);
      pack_weight_kernel<<<(waves + 3) / 4, 128, 0, stream>>>(s, d, K, N);
    };
    packs(W_m1,   pkWm1,   64,  128);
    packs(W_m2,   pkWm2,   128, 32);
    packs(W_lin2, pkWlin2, 32,  64);
    packs(W_lin1, pkWlin1, 32,  64);
    packs(W1,     pkW1,    128, 256);
    packs(W2,     pkW2,    256, 256);
    packs(W3,     pkW3,    256, 16);
  }
  // 4) edge MLP + scatter (1M edges, 16 per wave, 4 waves/block)
  edge_kernel<<<NE / (4 * 16), 128, 0, stream>>>(convs, edges, pkWm1, pkWm2, pkWlin2,
                                                 b_m1, b_m2, b_lin2, agg);
  // 5) fused preconv + 3-layer MLP (16 rows/wave, 2 waves/block)
  {
    int tiles  = (NKEEP + 15) / 16;      // 16257
    int blocks = (tiles + 1) / 2;        // 8129
    node_mlp_kernel<<<blocks, 64, 0, stream>>>(features, convsbf, agg,
                                               pkWlin1, b_lin1, pkW1, b1,
                                               pkW2, b2, pkW3, b3, out);
  }
}
